// MVPool_GCN_60413009985911
// MI455X (gfx1250) — matrix operand order repaired and checked by reading, not compiled, not run on hardware
//
#include <hip/hip_runtime.h>
#include <hip/hip_bf16.h>

// ---------------- problem constants (match reference) ----------------
#define NNODES 50000
#define NEDGES 800000
#define FEAT   128
#define KP1    40000
#define KP2    32000
#define PR_ITERS 10
#define DAMP   0.85f

typedef __attribute__((ext_vector_type(16))) _Float16 v16h;
typedef __attribute__((ext_vector_type(8)))  float    v8f;

__device__ __forceinline__ float sigmf(float v) { return 1.f / (1.f + expf(-v)); }

// ---------------- utility fills ----------------
__global__ void mvg_fill_f32(float* p, float v, int n) {
  int i = blockIdx.x * blockDim.x + threadIdx.x;
  if (i < n) p[i] = v;
}
__global__ void mvg_fill_u32(unsigned* p, unsigned v, int n) {
  int i = blockIdx.x * blockDim.x + threadIdx.x;
  if (i < n) p[i] = v;
}

// ---------------- standard scaler ----------------
__global__ void mvg_colstats(const float* __restrict__ x, float* mu, float* sd, int n) {
  int f = blockIdx.x;  // FEAT blocks
  float s = 0.f, s2 = 0.f;
  for (int r = threadIdx.x; r < n; r += blockDim.x) {
    float v = x[r * FEAT + f]; s += v; s2 += v * v;
  }
  __shared__ float a0[256], a1[256];
  a0[threadIdx.x] = s; a1[threadIdx.x] = s2; __syncthreads();
  for (int o = 128; o > 0; o >>= 1) {
    if (threadIdx.x < o) { a0[threadIdx.x] += a0[threadIdx.x + o]; a1[threadIdx.x] += a1[threadIdx.x + o]; }
    __syncthreads();
  }
  if (threadIdx.x == 0) {
    float m = a0[0] / (float)n;
    mu[f] = m;
    sd[f] = sqrtf(fmaxf(a1[0] / (float)n - m * m, 0.f));
  }
}
__global__ void mvg_scale(const float* __restrict__ x, const float* mu, const float* sd,
                          float* out, int total) {
  int i = blockIdx.x * blockDim.x + threadIdx.x;
  if (i >= total) return;
  int f = i & (FEAT - 1);
  out[i] = (x[i] - mu[f]) / fmaxf(sd[f], 1e-12f);
}

// ---------------- edge init ----------------
__global__ void mvg_edge_init(const int* __restrict__ ei, int* esrc, int* edst, float* ev, int ne) {
  int e = blockIdx.x * blockDim.x + threadIdx.x;
  if (e >= ne) return;
  esrc[e] = ei[e];
  edst[e] = ei[ne + e];
  ev[e] = 1.f;
}

// ---------------- f16 conversion ----------------
__global__ void mvg_cvt_f16(const float* __restrict__ a, _Float16* __restrict__ o, int total) {
  int i = blockIdx.x * blockDim.x + threadIdx.x;
  if (i < total) o[i] = (_Float16)a[i];
}
// transpose W (FEATxFEAT) into col-major f16: o[j*FEAT+k] = W[k*FEAT+j]
__global__ void mvg_cvt_wT_f16(const float* __restrict__ W, _Float16* __restrict__ o) {
  int i = blockIdx.x * blockDim.x + threadIdx.x;
  if (i >= FEAT * FEAT) return;
  int k = i >> 7, j = i & 127;
  o[j * FEAT + k] = (_Float16)W[k * FEAT + j];
}

// ---------------- WMMA GEMM: C[n x 128] = A[n x 128] * W  (Bt = W^T, col-major) ----------
// one wave per 16x16 output tile; block = 256 threads = 8 waves = one 16-row x 128-col strip
__global__ __launch_bounds__(256) void mvg_gemm_wmma(const _Float16* __restrict__ A,
                                                     const _Float16* __restrict__ Bt,
                                                     float* __restrict__ C) {
  const int wave = threadIdx.x >> 5;
  const int lane = threadIdx.x & 31;
  const int hi   = lane >> 4;        // which half-wave
  const int trow = blockIdx.x;       // 16-row tile index
  const int row  = trow * 16 + (lane & 15);   // A-matrix M for this lane
  const int col  = wave * 16 + (lane & 15);   // B/C/D N for this lane

  __builtin_prefetch(&A[row * FEAT], 0, 1);   // global_prefetch_b8

  v8f acc = {};
#pragma unroll
  for (int kk = 0; kk < FEAT; kk += 32) {
    v16h a, b;
#pragma unroll
    for (int v = 0; v < 8; ++v) {
      // A 16x32 f16 layout: lanes 0-15: VGPR0..3 -> K 0..7, VGPR4..7 -> K 16..23;
      //                     lanes 16-31 shifted by +8 within each 16-group.
      int ka = kk + ((v >= 4) ? 16 : 0) + hi * 8 + (v & 3) * 2;
      // B 32x16 f16 layout: same K order as A (lanes 0-15: K 0..7,16..23; lanes 16-31: K 8..15,24..31).
      int kb = kk + ((v >= 4) ? 16 : 0) + hi * 8 + (v & 3) * 2;
      a[2 * v]     = A[row * FEAT + ka];
      a[2 * v + 1] = A[row * FEAT + ka + 1];
      b[2 * v]     = Bt[col * FEAT + kb];
      b[2 * v + 1] = Bt[col * FEAT + kb + 1];
    }
    acc = __builtin_amdgcn_wmma_f32_16x16x32_f16(false, a, false, b, (short)0, acc, false, false);
  }
#pragma unroll
  for (int r = 0; r < 8; ++r) {
    int m = r + hi * 8;   // C/D layout: VGPR r holds row M=r (lanes 0-15) / M=r+8 (lanes 16-31)
    C[(trow * 16 + m) * FEAT + col] = acc[r];
  }
}

// ---------------- GCN conv pieces ----------------
__global__ void mvg_deg_scatter(const int* __restrict__ idx, const float* __restrict__ ev,
                                float* deg, int ne) {
  int e = blockIdx.x * blockDim.x + threadIdx.x;
  if (e >= ne) return;
  float v = ev[e];
  if (v != 0.f) atomicAdd(&deg[idx[e]], v);
}
__global__ void mvg_gcn_agg(const int* __restrict__ src, const int* __restrict__ dst,
                            const float* __restrict__ ev, const float* __restrict__ deg,
                            const float* __restrict__ h, float* __restrict__ out) {
  int e = blockIdx.x;          // one block per edge
  int f = threadIdx.x;         // 128 features
  float v = ev[e];
  if (v == 0.f) return;
  int s = src[e], d = dst[e];
  float norm = rsqrtf(deg[s]) * rsqrtf(deg[d]) * v;
  atomicAdd(&out[d * FEAT + f], norm * h[s * FEAT + f]);
}
__global__ void mvg_gcn_finish(float* __restrict__ out, const float* __restrict__ h,
                               const float* __restrict__ deg, const float* __restrict__ b,
                               int total) {
  int i = blockIdx.x * blockDim.x + threadIdx.x;
  if (i >= total) return;
  int node = i >> 7, f = i & 127;
  float v = out[i] + h[i] / deg[node] + b[f];
  out[i] = fmaxf(v, 0.f);   // fused relu
}

// ---------------- MVPool scoring ----------------
__global__ void mvg_score1(const float* __restrict__ odeg, const float* alpha, const float* beta,
                           float* s1, int n) {
  int i = blockIdx.x * blockDim.x + threadIdx.x;
  if (i >= n) return;
  s1[i] = sigmf(alpha[0] * logf(odeg[i] + 1e-16f) + beta[0]);
}
__global__ void mvg_wnorm(const float* __restrict__ w, float* out) {  // 1 block, 128 thr
  __shared__ float s[128];
  float v = w[threadIdx.x];
  s[threadIdx.x] = v * v; __syncthreads();
  for (int o = 64; o > 0; o >>= 1) { if (threadIdx.x < o) s[threadIdx.x] += s[threadIdx.x + o]; __syncthreads(); }
  if (threadIdx.x == 0) out[0] = sqrtf(s[0]);
}
__global__ void mvg_score2(const float* __restrict__ x, const float* __restrict__ w,
                           const float* wnorm, float* s2, int n) {
  int node = blockIdx.x;   // grid n, block 128
  __shared__ float s[128];
  s[threadIdx.x] = x[node * FEAT + threadIdx.x] * w[threadIdx.x];
  __syncthreads();
  for (int o = 64; o > 0; o >>= 1) { if (threadIdx.x < o) s[threadIdx.x] += s[threadIdx.x + o]; __syncthreads(); }
  if (threadIdx.x == 0) s2[node] = sigmf(s[0] / wnorm[0]);
}
__global__ void mvg_pr_init(const float* __restrict__ odeg, float* pr, float* inv, float* dng, int n) {
  int i = blockIdx.x * blockDim.x + threadIdx.x;
  if (i >= n) return;
  float od = odeg[i];
  pr[i]  = 1.f / (float)n;
  inv[i] = (od > 0.f) ? 1.f / fmaxf(od, 1e-12f) : 0.f;
  dng[i] = (od <= 0.f) ? 1.f : 0.f;
}
__global__ void mvg_pr_edge(const int* __restrict__ src, const int* __restrict__ dst,
                            const float* __restrict__ ev, const float* __restrict__ pr,
                            const float* __restrict__ inv, float* ct, int ne) {
  int e = blockIdx.x * blockDim.x + threadIdx.x;
  if (e >= ne) return;
  float v = ev[e];
  if (v == 0.f) return;
  int s = src[e];
  atomicAdd(&ct[dst[e]], v * pr[s] * inv[s]);
}
__global__ void mvg_dang(const float* __restrict__ pr, const float* __restrict__ dng,
                         float* out, int n) {
  __shared__ float s[256];
  float acc = 0.f;
  for (int i = blockIdx.x * 256 + threadIdx.x; i < n; i += gridDim.x * 256) acc += pr[i] * dng[i];
  s[threadIdx.x] = acc; __syncthreads();
  for (int o = 128; o > 0; o >>= 1) { if (threadIdx.x < o) s[threadIdx.x] += s[threadIdx.x + o]; __syncthreads(); }
  if (threadIdx.x == 0) atomicAdd(out, s[0]);
}
__global__ void mvg_pr_update(float* pr, const float* __restrict__ ct, const float* dang, int n) {
  int i = blockIdx.x * blockDim.x + threadIdx.x;
  if (i >= n) return;
  pr[i] = (1.f - DAMP) / (float)n + DAMP * (ct[i] + dang[0] / (float)n);
}
__global__ void mvg_score3(const float* __restrict__ pr, float* s3, int n) {
  int i = blockIdx.x * blockDim.x + threadIdx.x;
  if (i < n) s3[i] = sigmf(pr[i]);
}
__global__ void mvg_maxabs(const float* s1, const float* s2, const float* s3, unsigned* mx, int n) {
  int i = blockIdx.x * blockDim.x + threadIdx.x;
  if (i >= n) return;
  // sigmoid outputs are positive -> IEEE bit pattern is order-preserving
  atomicMax(&mx[0], __float_as_uint(s1[i]));
  atomicMax(&mx[1], __float_as_uint(s2[i]));
  atomicMax(&mx[2], __float_as_uint(s3[i]));
}
__global__ void mvg_fuse(const float* s1, const float* s2, const float* s3, const unsigned* mx,
                         const float* __restrict__ vatt, const float* __restrict__ vbias,
                         float* score, unsigned* keys, int n) {
  int i = blockIdx.x * blockDim.x + threadIdx.x;
  if (i >= n) return;
  float sc[3];
  sc[0] = s1[i] / __uint_as_float(mx[0]);
  sc[1] = s2[i] / __uint_as_float(mx[1]);
  sc[2] = s3[i] / __uint_as_float(mx[2]);
  float z[3], zmax = -1e30f;
  for (int j = 0; j < 3; ++j) {
    float t = sc[0] * vatt[0 * 3 + j] + sc[1] * vatt[1 * 3 + j] + sc[2] * vatt[2 * 3 + j] + vbias[j];
    z[j] = sigmf(t);
    zmax = fmaxf(zmax, z[j]);
  }
  float den = 0.f, num = 0.f;
  for (int j = 0; j < 3; ++j) { float e = expf(z[j] - zmax); den += e; num += sc[j] * e; }
  float s = sigmf(num / den);
  score[i] = s;
  keys[i] = __float_as_uint(s);
}

// ---------------- radix select (k-th largest) ----------------
// STATE[0]=prefix/threshold, STATE[1]=remaining (ties needed after all passes)
__global__ void mvg_sel_init(unsigned* st, int k) {
  if (threadIdx.x == 0) { st[0] = 0u; st[1] = (unsigned)k; }
}
__global__ void mvg_hist(const unsigned* __restrict__ keys, const unsigned* st, unsigned* hist,
                         unsigned himask, int shift, int n) {
  int i = blockIdx.x * blockDim.x + threadIdx.x;
  if (i >= n) return;
  unsigned key = keys[i];
  if ((key & himask) == (st[0] & himask)) atomicAdd(&hist[(key >> shift) & 255u], 1u);
}
__global__ void mvg_sel_digit(const unsigned* __restrict__ hist, unsigned* st, int shift) {
  if (threadIdx.x != 0) return;
  unsigned rem = st[1], cum = 0u;
  for (int d = 255; d >= 0; --d) {
    unsigned c = hist[d];
    if (cum + c >= rem) { st[0] |= ((unsigned)d) << shift; st[1] = rem - cum; return; }
    cum += c;
  }
}
__global__ void mvg_mark_tie(const unsigned* keys, const unsigned* st, unsigned* flg, int n) {
  int i = blockIdx.x * blockDim.x + threadIdx.x;
  if (i < n) flg[i] = (keys[i] == st[0]) ? 1u : 0u;
}
__global__ void mvg_mark_sel(const unsigned* keys, const unsigned* st, const unsigned* tierank,
                             unsigned* flg, int n) {
  int i = blockIdx.x * blockDim.x + threadIdx.x;
  if (i >= n) return;
  unsigned key = keys[i];
  bool sel = (key > st[0]) || (key == st[0] && tierank[i] < st[1]);
  flg[i] = sel ? 1u : 0u;
}

// ---------------- exclusive scan (n <= 256*256) ----------------
__global__ void mvg_scan_block(const unsigned* __restrict__ flg, unsigned* out,
                               unsigned* bsum, int n) {
  __shared__ unsigned tmp[256];
  int i = blockIdx.x * 256 + threadIdx.x;
  unsigned v = (i < n) ? flg[i] : 0u;
  tmp[threadIdx.x] = v; __syncthreads();
  for (int off = 1; off < 256; off <<= 1) {
    unsigned t = (threadIdx.x >= off) ? tmp[threadIdx.x - off] : 0u;
    __syncthreads();
    tmp[threadIdx.x] += t;
    __syncthreads();
  }
  if (i < n) out[i] = tmp[threadIdx.x] - v;
  if (threadIdx.x == 255) bsum[blockIdx.x] = tmp[255];
}
__global__ void mvg_scan_sums(unsigned* bsum, int nb) {  // 1 block, nb <= 256
  __shared__ unsigned tmp[256];
  unsigned v = (threadIdx.x < nb) ? bsum[threadIdx.x] : 0u;
  tmp[threadIdx.x] = v; __syncthreads();
  for (int off = 1; off < 256; off <<= 1) {
    unsigned t = (threadIdx.x >= off) ? tmp[threadIdx.x - off] : 0u;
    __syncthreads();
    tmp[threadIdx.x] += t;
    __syncthreads();
  }
  if (threadIdx.x < nb) bsum[threadIdx.x] = tmp[threadIdx.x] - v;
}
__global__ void mvg_scan_add(unsigned* out, const unsigned* bsum, int n) {
  int i = blockIdx.x * 256 + threadIdx.x;
  if (i < n) out[i] += bsum[blockIdx.x];
}

// ---------------- compaction / relabel ----------------
__global__ void mvg_write_newidx(const unsigned* flg, const unsigned* rank, int* nidx, int n) {
  int i = blockIdx.x * blockDim.x + threadIdx.x;
  if (i < n) nidx[i] = flg[i] ? (int)rank[i] : -1;
}
__global__ void mvg_pool_gather(const float* __restrict__ x, const float* __restrict__ score,
                                const int* __restrict__ nidx, float* __restrict__ out) {
  int node = blockIdx.x;
  int d = nidx[node];
  if (d < 0) return;
  out[d * FEAT + threadIdx.x] = x[node * FEAT + threadIdx.x] * score[node];
}
__global__ void mvg_edge_relabel(int* src, int* dst, float* ev, const int* __restrict__ nidx, int ne) {
  int e = blockIdx.x * blockDim.x + threadIdx.x;
  if (e >= ne) return;
  int ns = nidx[src[e]], nd = nidx[dst[e]];
  bool keep = (ns >= 0) && (nd >= 0);
  ev[e] = keep ? ev[e] : 0.f;
  src[e] = (ns >= 0) ? ns : 0;
  dst[e] = (nd >= 0) ? nd : 0;
}

// ---------------- global max/mean pool ----------------
__global__ void mvg_gpool(const float* __restrict__ x, float* out, int k) {
  int f = blockIdx.x;  // FEAT blocks
  float mx = -1e30f, sm = 0.f;
  for (int r = threadIdx.x; r < k; r += 256) {
    float v = x[r * FEAT + f];
    mx = fmaxf(mx, v); sm += v;
  }
  __shared__ float sx[256], ss[256];
  sx[threadIdx.x] = mx; ss[threadIdx.x] = sm; __syncthreads();
  for (int o = 128; o > 0; o >>= 1) {
    if (threadIdx.x < o) {
      sx[threadIdx.x] = fmaxf(sx[threadIdx.x], sx[threadIdx.x + o]);
      ss[threadIdx.x] += ss[threadIdx.x + o];
    }
    __syncthreads();
  }
  if (threadIdx.x == 0) { out[f] = sx[0]; out[FEAT + f] = ss[0] / (float)k; }
}

// ---------------- classifier head + log_softmax ----------------
__global__ void mvg_head(const float* __restrict__ x1, const float* __restrict__ x2,
                         const float* __restrict__ Wlin, const float* __restrict__ blin,
                         float* out) {
  __shared__ float v[2 * FEAT];
  __shared__ float logit[10];
  for (int f = threadIdx.x; f < 2 * FEAT; f += blockDim.x)
    v[f] = fmaxf(x1[f], 0.f) + fmaxf(x2[f], 0.f);
  __syncthreads();
  if (threadIdx.x < 10) {
    float acc = blin[threadIdx.x];
    for (int f = 0; f < 2 * FEAT; ++f) acc += v[f] * Wlin[f * 10 + threadIdx.x];
    logit[threadIdx.x] = fmaxf(acc, 0.f);
  }
  __syncthreads();
  if (threadIdx.x == 0) {
    float m = -1e30f;
    for (int o = 0; o < 10; ++o) m = fmaxf(m, logit[o]);
    float den = 0.f;
    for (int o = 0; o < 10; ++o) den += expf(logit[o] - m);
    float lse = m + logf(den);
    for (int o = 0; o < 10; ++o) out[o] = logit[o] - lse;
  }
}

// =====================================================================
extern "C" void kernel_launch(void* const* d_in, const int* in_sizes, int n_in,
                              void* d_out, int out_size, void* d_ws, size_t ws_size,
                              hipStream_t stream) {
  (void)in_sizes; (void)n_in; (void)out_size;
  const float* x     = (const float*)d_in[0];
  const int*   ei    = (const int*)d_in[1];
  const float* W1    = (const float*)d_in[2];
  const float* b1    = (const float*)d_in[3];
  const float* W2    = (const float*)d_in[4];
  const float* b2    = (const float*)d_in[5];
  const float* Wlin  = (const float*)d_in[6];
  const float* blin  = (const float*)d_in[7];
  const float* pw    = (const float*)d_in[8];
  const float* vatt  = (const float*)d_in[9];
  const float* vbias = (const float*)d_in[10];
  const float* alpha = (const float*)d_in[11];
  const float* beta  = (const float*)d_in[12];
  float* out = (float*)d_out;

  // -------- workspace carve-up --------
  size_t off = 0;
  auto alloc = [&](size_t bytes) -> void* {
    void* p = (char*)d_ws + off;
    off += (bytes + 255) & ~(size_t)255;
    return p;
  };
  float*     XA    = (float*)alloc((size_t)NNODES * FEAT * 4);
  float*     XB    = (float*)alloc((size_t)NNODES * FEAT * 4);
  float*     HH    = (float*)alloc((size_t)NNODES * FEAT * 4);
  _Float16*  XH    = (_Float16*)alloc((size_t)NNODES * FEAT * 2);
  _Float16*  WH    = (_Float16*)alloc((size_t)FEAT * FEAT * 2);
  float*     DEG   = (float*)alloc((size_t)NNODES * 4);
  float*     ODEG  = (float*)alloc((size_t)NNODES * 4);
  float*     INV   = (float*)alloc((size_t)NNODES * 4);
  float*     DNG   = (float*)alloc((size_t)NNODES * 4);
  float*     PR    = (float*)alloc((size_t)NNODES * 4);
  float*     CT    = (float*)alloc((size_t)NNODES * 4);
  float*     S1    = (float*)alloc((size_t)NNODES * 4);
  float*     S2    = (float*)alloc((size_t)NNODES * 4);
  float*     S3    = (float*)alloc((size_t)NNODES * 4);
  float*     SCORE = (float*)alloc((size_t)NNODES * 4);
  unsigned*  KEYS  = (unsigned*)alloc((size_t)NNODES * 4);
  unsigned*  TRANK = (unsigned*)alloc((size_t)NNODES * 4);
  unsigned*  FLG   = (unsigned*)alloc((size_t)NNODES * 4);
  int*       NIDX  = (int*)alloc((size_t)NNODES * 4);
  unsigned*  BSUM  = (unsigned*)alloc(256 * 4);
  int*       ESRC  = (int*)alloc((size_t)NEDGES * 4);
  int*       EDST  = (int*)alloc((size_t)NEDGES * 4);
  float*     EV    = (float*)alloc((size_t)NEDGES * 4);
  float*     MU    = (float*)alloc(FEAT * 4);
  float*     SD    = (float*)alloc(FEAT * 4);
  float*     X1v   = (float*)alloc(2 * FEAT * 4);
  float*     X2v   = (float*)alloc(2 * FEAT * 4);
  unsigned*  HIST  = (unsigned*)alloc(256 * 4);
  unsigned*  STATE = (unsigned*)alloc(16 * 4);
  unsigned*  MAXS  = (unsigned*)alloc(4 * 4);
  float*     DANG  = (float*)alloc(4);
  float*     WNORM = (float*)alloc(4);
  if (off > ws_size) return;  // workspace too small — bail safely

  auto g256 = [](int n) { return (n + 255) / 256; };
  auto fillf = [&](float* p, float v, int n) { mvg_fill_f32<<<g256(n), 256, 0, stream>>>(p, v, n); };
  auto fillu = [&](unsigned* p, unsigned v, int n) { mvg_fill_u32<<<g256(n), 256, 0, stream>>>(p, v, n); };
  auto scan = [&](unsigned* flg, unsigned* rank, int n) {
    int nb = g256(n);  // <= 256 for n <= 65536
    mvg_scan_block<<<nb, 256, 0, stream>>>(flg, rank, BSUM, n);
    mvg_scan_sums<<<1, 256, 0, stream>>>(BSUM, nb);
    mvg_scan_add<<<nb, 256, 0, stream>>>(rank, BSUM, n);
  };

  auto conv = [&](const float* xin, float* xout, int n, const float* W, const float* b) {
    mvg_cvt_f16<<<g256(n * FEAT), 256, 0, stream>>>(xin, XH, n * FEAT);
    mvg_cvt_wT_f16<<<g256(FEAT * FEAT), 256, 0, stream>>>(W, WH);
    mvg_gemm_wmma<<<n / 16, 256, 0, stream>>>(XH, WH, HH);
    fillf(DEG, 1.f, n);  // self-loop +1
    mvg_deg_scatter<<<g256(NEDGES), 256, 0, stream>>>(EDST, EV, DEG, NEDGES);
    fillf(xout, 0.f, n * FEAT);
    mvg_gcn_agg<<<NEDGES, FEAT, 0, stream>>>(ESRC, EDST, EV, DEG, HH, xout);
    mvg_gcn_finish<<<g256(n * FEAT), 256, 0, stream>>>(xout, HH, DEG, b, n * FEAT);
  };

  auto pool = [&](const float* xin, float* xout, int n, int k) {
    fillf(ODEG, 0.f, n);
    mvg_deg_scatter<<<g256(NEDGES), 256, 0, stream>>>(ESRC, EV, ODEG, NEDGES);
    mvg_score1<<<g256(n), 256, 0, stream>>>(ODEG, alpha, beta, S1, n);
    mvg_score2<<<n, 128, 0, stream>>>(xin, pw, WNORM, S2, n);
    mvg_pr_init<<<g256(n), 256, 0, stream>>>(ODEG, PR, INV, DNG, n);
    for (int it = 0; it < PR_ITERS; ++it) {
      fillf(CT, 0.f, n);
      mvg_pr_edge<<<g256(NEDGES), 256, 0, stream>>>(ESRC, EDST, EV, PR, INV, CT, NEDGES);
      fillf(DANG, 0.f, 1);
      mvg_dang<<<64, 256, 0, stream>>>(PR, DNG, DANG, n);
      mvg_pr_update<<<g256(n), 256, 0, stream>>>(PR, CT, DANG, n);
    }
    mvg_score3<<<g256(n), 256, 0, stream>>>(PR, S3, n);
    fillu(MAXS, 0u, 3);
    mvg_maxabs<<<g256(n), 256, 0, stream>>>(S1, S2, S3, MAXS, n);
    mvg_fuse<<<g256(n), 256, 0, stream>>>(S1, S2, S3, MAXS, vatt, vbias, SCORE, KEYS, n);
    // radix select k-th largest score
    mvg_sel_init<<<1, 1, 0, stream>>>(STATE, k);
    for (int pass = 3; pass >= 0; --pass) {
      int shift = pass * 8;
      unsigned himask = (pass == 3) ? 0u : (0xFFFFFFFFu << (shift + 8));
      fillu(HIST, 0u, 256);
      mvg_hist<<<g256(n), 256, 0, stream>>>(KEYS, STATE, HIST, himask, shift, n);
      mvg_sel_digit<<<1, 1, 0, stream>>>(HIST, STATE, shift);
    }
    // stable compaction: ties first, then full selection
    mvg_mark_tie<<<g256(n), 256, 0, stream>>>(KEYS, STATE, FLG, n);
    scan(FLG, TRANK, n);
    mvg_mark_sel<<<g256(n), 256, 0, stream>>>(KEYS, STATE, TRANK, FLG, n);
    scan(FLG, TRANK, n);
    mvg_write_newidx<<<g256(n), 256, 0, stream>>>(FLG, TRANK, NIDX, n);
    mvg_pool_gather<<<n, FEAT, 0, stream>>>(xin, SCORE, NIDX, xout);
    mvg_edge_relabel<<<g256(NEDGES), 256, 0, stream>>>(ESRC, EDST, EV, NIDX, NEDGES);
  };

  // -------- pipeline --------
  mvg_colstats<<<FEAT, 256, 0, stream>>>(x, MU, SD, NNODES);
  mvg_scale<<<g256(NNODES * FEAT), 256, 0, stream>>>(x, MU, SD, XA, NNODES * FEAT);
  mvg_edge_init<<<g256(NEDGES), 256, 0, stream>>>(ei, ESRC, EDST, EV, NEDGES);
  mvg_wnorm<<<1, 128, 0, stream>>>(pw, WNORM);

  conv(XA, XB, NNODES, W1, b1);          // relu(gcn_conv 1)
  pool(XB, XA, NNODES, KP1);             // mvpool 1 -> XA[0:K1]
  mvg_gpool<<<FEAT, 256, 0, stream>>>(XA, X1v, KP1);

  conv(XA, XB, KP1, W2, b2);             // relu(gcn_conv 2)
  pool(XB, XA, KP1, KP2);                // mvpool 2 -> XA[0:K2]
  mvg_gpool<<<FEAT, 256, 0, stream>>>(XA, X2v, KP2);

  mvg_head<<<1, 64, 0, stream>>>(X1v, X2v, Wlin, blin, out);
}
